// SSDenseTransMoEBlock_49443663512209
// MI455X (gfx1250) — matrix operand order repaired and checked by reading, not compiled, not run on hardware
//
#include <hip/hip_runtime.h>

// ---------------- problem constants ----------------
#define NTOK   2048          // 2 * 1024 tokens
#define FDIM   2048          // model dim
#define HDIM   4096          // expert hidden dim
#define NEXP   8
#define TOPK   2
#define PAIRS  (NTOK * TOPK)               // 4096
#define MTILE  128                         // workgroup M tile (4 waves of 32)
#define NTILE  64                          // workgroup N tile (2 waves of 32)
#define KSTEP  32
#define LDA    40                          // LDS row stride in bf16 elems (80B: 16B-aligned, conflict-free b128 reads)
#define LDAU   5                           // row stride in 16B units
#define LDAW   20                          // row stride in u32 words
#define MAXSLOTS (PAIRS + NEXP * MTILE)    // 5120 (128-aligned per-expert segments)
#define MAXTILES (MAXSLOTS / MTILE)        // 40

// ---------------- CDNA5 WMMA types ----------------
typedef __attribute__((ext_vector_type(16))) __bf16 v16bf;
typedef __attribute__((ext_vector_type(8)))  float  v8f;
typedef __attribute__((ext_vector_type(4)))  int    v4i;

union Frag16 { v16bf v; v4i q[2]; };

// Direct packed fp32->bf16x2 convert: D[15:0]=bf16(lo), D[31:16]=bf16(hi).
static __device__ __forceinline__ unsigned int pack2_bf16(float lo, float hi) {
  unsigned int r;
  asm("v_cvt_pk_bf16_f32 %0, %1, %2" : "=v"(r) : "v"(lo), "v"(hi));
  return r;
}

// ---------------- 1. fp32 -> bf16 activation convert ----------------
__global__ __launch_bounds__(256) void convert_kernel(const float* __restrict__ x,
                                                      __bf16* __restrict__ xb) {
  size_t i = ((size_t)blockIdx.x * 256 + threadIdx.x);   // float4 units
  const float4 v = ((const float4*)x)[i];
  uint2 o;
  o.x = pack2_bf16(v.x, v.y);
  o.y = pack2_bf16(v.z, v.w);
  ((uint2*)xb)[i] = o;
}

// ---------------- 2. router: logits, softmax, top-2, renorm ----------------
__global__ __launch_bounds__(256) void router_kernel(const float* __restrict__ x,
                                                     const float* __restrict__ gate_w,
                                                     int* __restrict__ topk_idx,
                                                     float* __restrict__ topk_w) {
  __shared__ float red[256][NEXP];
  const int n = blockIdx.x, tid = threadIdx.x;
  float acc[NEXP];
#pragma unroll
  for (int e = 0; e < NEXP; ++e) acc[e] = 0.f;
#pragma unroll
  for (int i = 0; i < FDIM / 256; ++i) {
    int f = tid + i * 256;
    float xv = x[(size_t)n * FDIM + f];
#pragma unroll
    for (int e = 0; e < NEXP; ++e) acc[e] += xv * gate_w[e * FDIM + f];
  }
#pragma unroll
  for (int e = 0; e < NEXP; ++e) red[tid][e] = acc[e];
  __syncthreads();
  for (int s = 128; s > 0; s >>= 1) {
    if (tid < s) {
#pragma unroll
      for (int e = 0; e < NEXP; ++e) red[tid][e] += red[tid + s][e];
    }
    __syncthreads();
  }
  if (tid == 0) {
    float l[NEXP], p[NEXP];
    float m = red[0][0];
#pragma unroll
    for (int e = 0; e < NEXP; ++e) { l[e] = red[0][e]; m = fmaxf(m, l[e]); }
    float sum = 0.f;
#pragma unroll
    for (int e = 0; e < NEXP; ++e) { p[e] = __expf(l[e] - m); sum += p[e]; }
    int i0 = 0;
#pragma unroll
    for (int e = 1; e < NEXP; ++e) if (p[e] > p[i0]) i0 = e;
    int i1 = (i0 == 0) ? 1 : 0;
#pragma unroll
    for (int e = 0; e < NEXP; ++e) if (e != i0 && p[e] > p[i1]) i1 = e;
    float g0 = p[i0] / sum, g1 = p[i1] / sum;
    float gs = g0 + g1;
    topk_idx[2 * n]     = i0;
    topk_idx[2 * n + 1] = i1;
    topk_w[2 * n]       = g0 / gs;
    topk_w[2 * n + 1]   = g1 / gs;
  }
}

// ---------------- 3. per-expert pair counts (single block, int atomics) ----------------
__global__ __launch_bounds__(256) void counts_kernel(const int* __restrict__ topk_idx,
                                                     int* __restrict__ counts) {
  __shared__ int c[NEXP];
  int tid = threadIdx.x;
  if (tid < NEXP) c[tid] = 0;
  __syncthreads();
  for (int p = tid; p < PAIRS; p += 256) atomicAdd(&c[topk_idx[p]], 1);
  __syncthreads();
  if (tid < NEXP) counts[tid] = c[tid];
}

// ---------------- 4. 128-aligned segment offsets ----------------
__global__ void offsets_kernel(const int* __restrict__ counts, int* __restrict__ aoff) {
  if (threadIdx.x == 0 && blockIdx.x == 0) {
    int acc = 0;
#pragma unroll
    for (int e = 0; e < NEXP; ++e) { aoff[e] = acc; acc += (counts[e] + MTILE - 1) & ~(MTILE - 1); }
    aoff[NEXP] = acc;
  }
}

// ---------------- 5. stable compaction: pair -> slot (deterministic prefix scan) ----------------
__global__ __launch_bounds__(256) void compact_kernel(const int* __restrict__ topk_idx,
                                                      const int* __restrict__ aoff,
                                                      int* __restrict__ token_of_slot,
                                                      int* __restrict__ pos_of_pair) {
  __shared__ int sc[256];
  const int e = blockIdx.x, tid = threadIdx.x;
  const int base = aoff[e];
  int runbase = base;
  for (int chunk = 0; chunk < PAIRS; chunk += 256) {
    int p = chunk + tid;
    int flag = (topk_idx[p] == e) ? 1 : 0;
    sc[tid] = flag;
    __syncthreads();
    for (int s = 1; s < 256; s <<= 1) {
      int v = (tid >= s) ? sc[tid - s] : 0;
      __syncthreads();
      sc[tid] += v;
      __syncthreads();
    }
    int incl = sc[tid];
    int total = sc[255];
    if (flag) {
      int slot = runbase + incl - 1;
      token_of_slot[slot] = p >> 1;     // token id
      pos_of_pair[p] = slot;
    }
    runbase += total;
    __syncthreads();
  }
  int aend = aoff[e + 1];
  for (int s = runbase + tid; s < aend; s += 256) token_of_slot[s] = -1;   // pad rows
}

// ============================================================================
// 6. up-proj: h = silu(x@w1) * (x@w3)
//    128x64 workgroup tile, 8 waves (4 M x 2 N), 32x32 per wave,
//    register-staged ping-pong LDS pipeline, one barrier / K-step,
//    80B LDS row stride (conflict-free ds_load_b128 fragments).
// ============================================================================
__global__ __launch_bounds__(256) void moe_up_kernel(const __bf16* __restrict__ xb,
                                                     const float* __restrict__ w1,
                                                     const float* __restrict__ w3,
                                                     const int* __restrict__ aoff,
                                                     const int* __restrict__ token_of_slot,
                                                     __bf16* __restrict__ h_buf) {
  __shared__ __attribute__((aligned(16))) __bf16 As[2][MTILE * LDA];   // 10 KB x2
  __shared__ __attribute__((aligned(16))) __bf16 B1s[2][NTILE * LDA];  // 5 KB x2
  __shared__ __attribute__((aligned(16))) __bf16 B3s[2][NTILE * LDA];  // 5 KB x2

  const int tid = threadIdx.x;
  const int tile = blockIdx.x;
  const int total = aoff[NEXP];
  if (tile * MTILE >= total) return;          // uniform exit
  const int slot0 = tile * MTILE;
  int e = 0;
#pragma unroll
  for (int i = 0; i < NEXP; ++i) if (slot0 >= aoff[i + 1]) e = i + 1;
  const int n0 = blockIdx.y * NTILE;

  const float* w1p = w1 + (size_t)e * FDIM * HDIM;
  const float* w3p = w3 + (size_t)e * FDIM * HDIM;

  // ---- loop-invariant staging descriptors ----
  // A: 128 rows x 32 bf16 (data), thread -> rows (tid>>2), 64+(tid>>2), 16B chunk q = tid&3
  const int ra = tid >> 2;
  const int rb = 64 + ra;
  const int qa = tid & 3;
  const int ta = token_of_slot[slot0 + ra];
  const int tb = token_of_slot[slot0 + rb];
  const __bf16* pa = xb + (size_t)(ta < 0 ? 0 : ta) * FDIM + qa * 8;
  const __bf16* pb = xb + (size_t)(tb < 0 ? 0 : tb) * FDIM + qa * 8;
  // B: 32 rows(k) x 64 cols fp32; thread -> row-pair kk = (tid>>4)*2, float4 col nq = (tid&15)*4
  const int kk = (tid >> 4) * 2;
  const int nq = (tid & 15) * 4;
  const float* w1a = w1p + (size_t)kk * HDIM + n0 + nq;
  const float* w3a = w3p + (size_t)kk * HDIM + n0 + nq;

  // wave geometry
  const int lane = tid & 31;
  const int wv = tid >> 5;
  const int wm = (wv >> 1) * 32;        // wave M offset (0,32,64,96)
  const int wn = (wv & 1) * 32;         // wave N offset (0,32)
  const int m = lane & 15;
  const int half = lane >> 4;

  v8f acc1[2][2] = {}, acc3[2][2] = {};
  const v4i vzero = {0, 0, 0, 0};

  auto stage_store = [&](int p, v4i a0, v4i a1,
                         float4 x1a, float4 x1b, float4 x3a, float4 x3b) {
    ((v4i*)As[p])[ra * LDAU + qa] = a0;
    ((v4i*)As[p])[rb * LDAU + qa] = a1;
    unsigned int* d1 = (unsigned int*)B1s[p];
    unsigned int* d3 = (unsigned int*)B3s[p];
    const int kw = kk >> 1;
    d1[(nq + 0) * LDAW + kw] = pack2_bf16(x1a.x, x1b.x);
    d1[(nq + 1) * LDAW + kw] = pack2_bf16(x1a.y, x1b.y);
    d1[(nq + 2) * LDAW + kw] = pack2_bf16(x1a.z, x1b.z);
    d1[(nq + 3) * LDAW + kw] = pack2_bf16(x1a.w, x1b.w);
    d3[(nq + 0) * LDAW + kw] = pack2_bf16(x3a.x, x3b.x);
    d3[(nq + 1) * LDAW + kw] = pack2_bf16(x3a.y, x3b.y);
    d3[(nq + 2) * LDAW + kw] = pack2_bf16(x3a.z, x3b.z);
    d3[(nq + 3) * LDAW + kw] = pack2_bf16(x3a.w, x3b.w);
  };

  // ---- prologue: stage k0 = 0 into buffer 0 ----
  {
    v4i a0 = (ta >= 0) ? *(const v4i*)pa : vzero;
    v4i a1 = (tb >= 0) ? *(const v4i*)pb : vzero;
    float4 x1a = *(const float4*)(w1a);
    float4 x1b = *(const float4*)(w1a + HDIM);
    float4 x3a = *(const float4*)(w3a);
    float4 x3b = *(const float4*)(w3a + HDIM);
    stage_store(0, a0, a1, x1a, x1b, x3a, x3b);
  }
  __syncthreads();

  int p = 0;
  for (int k0 = 0; k0 < FDIM; k0 += KSTEP) {
    const int nk = k0 + KSTEP;
    const bool has_next = nk < FDIM;
    // --- issue next K-slab loads (overlap with WMMA below) ---
    v4i a0 = vzero, a1 = vzero;
    float4 x1a, x1b, x3a, x3b;
    if (has_next) {
      if (ta >= 0) a0 = *(const v4i*)(pa + nk);
      if (tb >= 0) a1 = *(const v4i*)(pb + nk);
      const size_t ko = (size_t)nk * HDIM;
      x1a = *(const float4*)(w1a + ko);
      x1b = *(const float4*)(w1a + ko + HDIM);
      x3a = *(const float4*)(w3a + ko);
      x3b = *(const float4*)(w3a + ko + HDIM);
    }
    // --- compute from buffer p ---
    const v4i* Aq  = (const v4i*)As[p];
    const v4i* B1q = (const v4i*)B1s[p];
    const v4i* B3q = (const v4i*)B3s[p];
    Frag16 af[2];
#pragma unroll
    for (int mb = 0; mb < 2; ++mb) {
      const int r = wm + mb * 16 + m;
      af[mb].q[0] = Aq[r * LDAU + half];
      af[mb].q[1] = Aq[r * LDAU + 2 + half];
    }
#pragma unroll
    for (int nb = 0; nb < 2; ++nb) {
      const int c = wn + nb * 16 + m;
      Frag16 b1f, b3f;
      b1f.q[0] = B1q[c * LDAU + half];
      b1f.q[1] = B1q[c * LDAU + 2 + half];
      b3f.q[0] = B3q[c * LDAU + half];
      b3f.q[1] = B3q[c * LDAU + 2 + half];
#pragma unroll
      for (int mb = 0; mb < 2; ++mb) {
        acc1[mb][nb] = __builtin_amdgcn_wmma_f32_16x16x32_bf16(
            false, af[mb].v, false, b1f.v, (short)0, acc1[mb][nb], false, false);
        acc3[mb][nb] = __builtin_amdgcn_wmma_f32_16x16x32_bf16(
            false, af[mb].v, false, b3f.v, (short)0, acc3[mb][nb], false, false);
      }
    }
    // --- store next slab into the other buffer ---
    if (has_next) stage_store(p ^ 1, a0, a1, x1a, x1b, x3a, x3b);
    __syncthreads();
    p ^= 1;
  }

  // ---- fused SiLU-gate epilogue; C/D layout: M = r + 8*half, N = lane&15 ----
#pragma unroll
  for (int nb = 0; nb < 2; ++nb) {
    const int outcol = n0 + wn + nb * 16 + m;
#pragma unroll
    for (int mb = 0; mb < 2; ++mb) {
      const int rowbase = slot0 + wm + mb * 16 + half * 8;
#pragma unroll
      for (int r = 0; r < 8; ++r) {
        float t1 = acc1[mb][nb][r], t3 = acc3[mb][nb][r];
        float hval = t1 * (1.0f / (1.0f + __expf(-t1))) * t3;
        h_buf[(size_t)(rowbase + r) * HDIM + outcol] = (__bf16)hval;
      }
    }
  }
}

// ============================================================================
// 7. down-proj: y = h @ w2   (same tiling, single B matrix)
// ============================================================================
__global__ __launch_bounds__(256) void moe_down_kernel(const __bf16* __restrict__ h_buf,
                                                       const float* __restrict__ w2,
                                                       const int* __restrict__ aoff,
                                                       float* __restrict__ y_pair) {
  __shared__ __attribute__((aligned(16))) __bf16 As[2][MTILE * LDA];   // 10 KB x2
  __shared__ __attribute__((aligned(16))) __bf16 Bs[2][NTILE * LDA];   // 5 KB x2

  const int tid = threadIdx.x;
  const int tile = blockIdx.x;
  const int total = aoff[NEXP];
  if (tile * MTILE >= total) return;
  const int slot0 = tile * MTILE;
  int e = 0;
#pragma unroll
  for (int i = 0; i < NEXP; ++i) if (slot0 >= aoff[i + 1]) e = i + 1;
  const int n0 = blockIdx.y * NTILE;

  const float* w2p = w2 + (size_t)e * HDIM * FDIM;

  const int ra = tid >> 2;
  const int rb = 64 + ra;
  const int qa = tid & 3;
  const __bf16* pa = h_buf + (size_t)(slot0 + ra) * HDIM + qa * 8;
  const __bf16* pb = h_buf + (size_t)(slot0 + rb) * HDIM + qa * 8;
  const int kk = (tid >> 4) * 2;
  const int nq = (tid & 15) * 4;
  const float* w2a = w2p + (size_t)kk * FDIM + n0 + nq;

  const int lane = tid & 31;
  const int wv = tid >> 5;
  const int wm = (wv >> 1) * 32;
  const int wn = (wv & 1) * 32;
  const int m = lane & 15;
  const int half = lane >> 4;

  v8f acc[2][2] = {};

  auto stage_store = [&](int p, v4i a0, v4i a1, float4 xa, float4 xb4) {
    ((v4i*)As[p])[ra * LDAU + qa] = a0;
    ((v4i*)As[p])[rb * LDAU + qa] = a1;
    unsigned int* d = (unsigned int*)Bs[p];
    const int kw = kk >> 1;
    d[(nq + 0) * LDAW + kw] = pack2_bf16(xa.x, xb4.x);
    d[(nq + 1) * LDAW + kw] = pack2_bf16(xa.y, xb4.y);
    d[(nq + 2) * LDAW + kw] = pack2_bf16(xa.z, xb4.z);
    d[(nq + 3) * LDAW + kw] = pack2_bf16(xa.w, xb4.w);
  };

  {
    v4i a0 = *(const v4i*)pa;
    v4i a1 = *(const v4i*)pb;
    float4 xa = *(const float4*)(w2a);
    float4 xb4 = *(const float4*)(w2a + FDIM);
    stage_store(0, a0, a1, xa, xb4);
  }
  __syncthreads();

  int p = 0;
  for (int k0 = 0; k0 < HDIM; k0 += KSTEP) {
    const int nk = k0 + KSTEP;
    const bool has_next = nk < HDIM;
    v4i a0, a1;
    float4 xa, xb4;
    if (has_next) {
      a0 = *(const v4i*)(pa + nk);
      a1 = *(const v4i*)(pb + nk);
      const size_t ko = (size_t)nk * FDIM;
      xa = *(const float4*)(w2a + ko);
      xb4 = *(const float4*)(w2a + ko + FDIM);
    }
    const v4i* Aq = (const v4i*)As[p];
    const v4i* Bq = (const v4i*)Bs[p];
    Frag16 af[2];
#pragma unroll
    for (int mb = 0; mb < 2; ++mb) {
      const int r = wm + mb * 16 + m;
      af[mb].q[0] = Aq[r * LDAU + half];
      af[mb].q[1] = Aq[r * LDAU + 2 + half];
    }
#pragma unroll
    for (int nb = 0; nb < 2; ++nb) {
      const int c = wn + nb * 16 + m;
      Frag16 bf;
      bf.q[0] = Bq[c * LDAU + half];
      bf.q[1] = Bq[c * LDAU + 2 + half];
#pragma unroll
      for (int mb = 0; mb < 2; ++mb) {
        acc[mb][nb] = __builtin_amdgcn_wmma_f32_16x16x32_bf16(
            false, af[mb].v, false, bf.v, (short)0, acc[mb][nb], false, false);
      }
    }
    if (has_next) stage_store(p ^ 1, a0, a1, xa, xb4);
    __syncthreads();
    p ^= 1;
  }

#pragma unroll
  for (int nb = 0; nb < 2; ++nb) {
    const int outcol = n0 + wn + nb * 16 + m;
#pragma unroll
    for (int mb = 0; mb < 2; ++mb) {
      const int rowbase = slot0 + wm + mb * 16 + half * 8;
#pragma unroll
      for (int r = 0; r < 8; ++r) {
        y_pair[(size_t)(rowbase + r) * FDIM + outcol] = acc[mb][nb][r];
      }
    }
  }
}

// ---------------- 8. combine: out[n] = g0*y[p0] + g1*y[p1] ----------------
__global__ __launch_bounds__(256) void combine_kernel(const float* __restrict__ y_pair,
                                                      const int* __restrict__ pos_of_pair,
                                                      const float* __restrict__ topk_w,
                                                      float* __restrict__ out) {
  const int idx = blockIdx.x * 256 + threadIdx.x;   // over NTOK * FDIM/4 float4 units
  const int n = idx / (FDIM / 4);
  const int c = idx % (FDIM / 4);
  const int p0 = pos_of_pair[2 * n];
  const int p1 = pos_of_pair[2 * n + 1];
  const float g0 = topk_w[2 * n];
  const float g1 = topk_w[2 * n + 1];
  const float4 a = ((const float4*)(y_pair + (size_t)p0 * FDIM))[c];
  const float4 b = ((const float4*)(y_pair + (size_t)p1 * FDIM))[c];
  float4 o;
  o.x = g0 * a.x + g1 * b.x;
  o.y = g0 * a.y + g1 * b.y;
  o.z = g0 * a.z + g1 * b.z;
  o.w = g0 * a.w + g1 * b.w;
  ((float4*)out)[idx] = o;
}

// ---------------- host launcher ----------------
extern "C" void kernel_launch(void* const* d_in, const int* in_sizes, int n_in,
                              void* d_out, int out_size, void* d_ws, size_t ws_size,
                              hipStream_t stream) {
  const float* x      = (const float*)d_in[0];
  const float* gate_w = (const float*)d_in[1];
  const float* w1     = (const float*)d_in[2];
  const float* w2     = (const float*)d_in[3];
  const float* w3     = (const float*)d_in[4];
  float* out = (float*)d_out;

  char* ws = (char*)d_ws;
  size_t off = 0;
  auto alloc = [&](size_t bytes) -> void* {
    void* p = ws + off;
    off = (off + bytes + 255) & ~(size_t)255;
    return p;
  };
  __bf16* xb            = (__bf16*)alloc((size_t)NTOK * FDIM * 2);
  int*    topk_idx      = (int*)   alloc((size_t)PAIRS * 4);
  float*  topk_w        = (float*) alloc((size_t)PAIRS * 4);
  int*    counts        = (int*)   alloc(NEXP * 4);
  int*    aoff          = (int*)   alloc((NEXP + 1) * 4);
  int*    pos_of_pair   = (int*)   alloc((size_t)PAIRS * 4);
  int*    token_of_slot = (int*)   alloc((size_t)MAXSLOTS * 4);
  __bf16* h_buf         = (__bf16*)alloc((size_t)MAXSLOTS * HDIM * 2);
  float*  y_pair        = (float*) alloc((size_t)MAXSLOTS * FDIM * 4);

  convert_kernel<<<(NTOK * FDIM / 4) / 256, 256, 0, stream>>>(x, xb);
  router_kernel<<<NTOK, 256, 0, stream>>>(x, gate_w, topk_idx, topk_w);
  counts_kernel<<<1, 256, 0, stream>>>(topk_idx, counts);
  offsets_kernel<<<1, 32, 0, stream>>>(counts, aoff);
  compact_kernel<<<NEXP, 256, 0, stream>>>(topk_idx, aoff, token_of_slot, pos_of_pair);
  moe_up_kernel<<<dim3(MAXTILES, HDIM / NTILE), 256, 0, stream>>>(xb, w1, w3, aoff, token_of_slot, h_buf);
  moe_down_kernel<<<dim3(MAXTILES, FDIM / NTILE), 256, 0, stream>>>(h_buf, w2, aoff, y_pair);
  combine_kernel<<<(NTOK * FDIM / 4) / 256, 256, 0, stream>>>(y_pair, pos_of_pair, topk_w, out);
}
